// BasicTransformer_16913581211726
// MI455X (gfx1250) — matrix operand order repaired and checked by reading, not compiled, not run on hardware
//
#include <hip/hip_runtime.h>

// ---------------------------------------------------------------------------
// MI455X (gfx1250) transformer forward. All GEMMs via v_wmma_f32_16x16x32_bf16,
// double-buffered LDS tiles staged with async global->LDS copies (ASYNCcnt),
// issued via inline asm (portable across ROCm/amdgpu-toolchain builtins).
// ---------------------------------------------------------------------------

typedef __attribute__((ext_vector_type(16))) __bf16 v16bf;
typedef __attribute__((ext_vector_type(8)))  float  v8f;

union FragAB { unsigned int u[8]; v16bf v; };

__device__ __forceinline__ unsigned short f2bf(float f) {
  unsigned int u = __float_as_uint(f);
  u += 0x7FFFu + ((u >> 16) & 1u);           // round-to-nearest-even
  return (unsigned short)(u >> 16);
}

static constexpr int TM   = 128;             // block tile M
static constexpr int TN   = 128;             // block tile N
static constexpr int TK   = 32;              // k step (WMMA bf16 K)
static constexpr int LDST = TK + 8;          // LDS row stride (elements): 80B rows

// ---- CDNA5 async global->LDS copy (inline asm; ASYNCcnt-tracked) ----------
#if defined(__AMDGCN__)
#define ASYNC_COPY 1
__device__ __forceinline__ void asyncCopy16B(const void* g, const void* l) {
  // Generic pointer to __shared__ is {aperture_hi32, lds_offset32}; the low
  // 32 bits are the wave-relative LDS byte address (ISA 10.2).
  const unsigned int lds = (unsigned int)(unsigned long long)(size_t)l;
  asm volatile("global_load_async_to_lds_b128 %0, %1, off"
               :: "v"(lds), "v"(g) : "memory");
}
__device__ __forceinline__ void waitAsync() {
#if __has_builtin(__builtin_amdgcn_s_wait_asynccnt)
  __builtin_amdgcn_s_wait_asynccnt(0);
#else
  asm volatile("s_wait_asynccnt 0" ::: "memory");
#endif
}
#else
#define ASYNC_COPY 0
#endif

// ---------------------------------------------------------------------------
// Generic bf16 WMMA GEMM: C[M,N] = A[M,K] * B
//   BNT=1 : B supplied as [N][K] row-major (pre-transposed / K-matrix rows)
//   BNT=0 : B supplied as [K][N] row-major (transposed into LDS on load)
//   OMODE=0 : fp32 out + bias      OMODE=1 : bf16 out
//   OMODE=2 : fp32 out, causal mask (col<=row ? val*scale : -inf), skip masked tiles
// Block: 256 threads = 8 waves (wave32). Wave grid 2(M)x4(N); each wave: 64x32 C.
// ---------------------------------------------------------------------------
template<int BNT, int OMODE>
__global__ void gemm_wmma(const unsigned short* __restrict__ A,
                          const unsigned short* __restrict__ B,
                          void* __restrict__ Cout,
                          const float* __restrict__ bias,
                          int M, int N, int K,
                          long sA, long sB, long sC,
                          float scale)
{
  __shared__ unsigned short As[2][TM * LDST];
  __shared__ unsigned short Bs[2][TN * LDST];

  const int tid  = threadIdx.x;
  const int lane = tid & 31;
  const int wave = tid >> 5;
  const int wm   = wave >> 2;                // 0..1  (64-row slab)
  const int wn   = wave & 3;                 // 0..3  (32-col slab)
  const int m0   = blockIdx.y * TM;
  const int n0   = blockIdx.x * TN;
  const int z    = blockIdx.z;

  if (OMODE == 2 && n0 > m0 + (TM - 1)) return;   // tile fully above diagonal

  A += (long)z * sA;
  B += (long)z * sB;

  v8f acc[4][2];
  const v8f vzero = {0.f,0.f,0.f,0.f,0.f,0.f,0.f,0.f};
#pragma unroll
  for (int i = 0; i < 4; ++i)
#pragma unroll
    for (int j = 0; j < 2; ++j) acc[i][j] = vzero;

  const int h   = lane >> 4;                 // lane half
  const int ln  = lane & 15;
  const int r   = tid >> 1;                  // staging row (2 threads/row)
  const int seg = (tid & 1) * 16;            // staging 16-element half

  // ---- tile staging (async when available) ----
  auto stageA = [&](int buf, int k0) {
    const unsigned short* src = A + (long)(m0 + r) * K + k0 + seg;
    unsigned short* dst = &As[buf][r * LDST + seg];
#if ASYNC_COPY
    asyncCopy16B(src, dst);
    asyncCopy16B(src + 8, dst + 8);
#else
    ((uint4*)dst)[0] = ((const uint4*)src)[0];
    ((uint4*)dst)[1] = ((const uint4*)src)[1];
#endif
  };
  auto stageB = [&](int buf, int k0) {
    if (BNT) {
      unsigned short* dst = &Bs[buf][r * LDST + seg];
      if (n0 + r < N) {
        const unsigned short* src = B + (long)(n0 + r) * K + k0 + seg;
#if ASYNC_COPY
        asyncCopy16B(src, dst);
        asyncCopy16B(src + 8, dst + 8);
#else
        ((uint4*)dst)[0] = ((const uint4*)src)[0];
        ((uint4*)dst)[1] = ((const uint4*)src)[1];
#endif
      } else {
        const uint4 zz = make_uint4(0u, 0u, 0u, 0u);
        ((uint4*)dst)[0] = zz;              // dscnt-tracked; drained by barrier
        ((uint4*)dst)[1] = zz;
      }
    } else {
      // transpose [K][N] tile -> LDS [n][k] (2-byte scatter; sync path)
      const int kk   = tid & 31;
      const int nseg = (tid >> 5) * 16;
#pragma unroll
      for (int j = 0; j < 16; ++j) {
        const int n = nseg + j;
        unsigned short val = 0;
        if (n0 + n < N) val = B[(long)(k0 + kk) * N + n0 + n];
        Bs[buf][n * LDST + kk] = val;
      }
    }
  };

  // ---- double-buffered main loop ----
  const int nSteps = K / TK;
  stageA(0, 0);
  stageB(0, 0);

  for (int s = 0; s < nSteps; ++s) {
    const int buf = s & 1;
#if ASYNC_COPY
    waitAsync();                             // own async stage for `buf` done
#endif
    __syncthreads();                         // all waves' stage of `buf` visible

    if (s + 1 < nSteps) {                    // prefetch next tile into other buffer
      stageA(buf ^ 1, (s + 1) * TK);
      stageB(buf ^ 1, (s + 1) * TK);
    }

    // ---- build fragments (ISA 7.12.2 layouts) and issue WMMAs ----
    v16bf af[4], bfrag[2];
#pragma unroll
    for (int mt = 0; mt < 4; ++mt) {
      const unsigned short* p = &As[buf][(wm * 64 + mt * 16 + ln) * LDST];
      FragAB f;
#pragma unroll
      for (int v = 0; v < 8; ++v) {
        const int klo = ((v < 4) ? 0 : 16) + h * 8 + (v & 3) * 2;
        f.u[v] = *(const unsigned int*)(p + klo);
      }
      af[mt] = f.v;
    }
#pragma unroll
    for (int nt = 0; nt < 2; ++nt) {
      const unsigned short* p = &Bs[buf][(wn * 32 + nt * 16 + ln) * LDST];
      FragAB f;
#pragma unroll
      for (int v = 0; v < 8; ++v) {
        const int klo = ((v < 4) ? 0 : 16) + h * 8 + (v & 3) * 2;
        f.u[v] = *(const unsigned int*)(p + klo);
      }
      bfrag[nt] = f.v;
    }
#pragma unroll
    for (int mt = 0; mt < 4; ++mt)
#pragma unroll
      for (int nt = 0; nt < 2; ++nt)
        acc[mt][nt] = __builtin_amdgcn_wmma_f32_16x16x32_bf16(
            false, af[mt], false, bfrag[nt], (short)0, acc[mt][nt], false, false);
  }

  // ---- store C: lane(h,ln) holds col=ln, rows = base + h*8 + v ----
#pragma unroll
  for (int mt = 0; mt < 4; ++mt) {
#pragma unroll
    for (int nt = 0; nt < 2; ++nt) {
      const int rbase = m0 + wm * 64 + mt * 16 + h * 8;
      const int col   = n0 + wn * 32 + nt * 16 + ln;
#pragma unroll
      for (int v = 0; v < 8; ++v) {
        const int row = rbase + v;
        const float val = acc[mt][nt][v];
        if (OMODE == 0) {
          if (col < N)
            ((float*)Cout)[(long)z * sC + (long)row * N + col] = val + bias[col];
        } else if (OMODE == 1) {
          if (col < N)
            ((unsigned short*)Cout)[(long)z * sC + (long)row * N + col] = f2bf(val);
        } else {
          const float o = (col <= row) ? val * scale : -__builtin_inff();
          ((float*)Cout)[(long)z * sC + (long)row * N + col] = o;
        }
      }
    }
  }
}

// ---------------------------------------------------------------------------
// x = W_embed[idx] + W_pos, stored as bf16
// ---------------------------------------------------------------------------
__global__ void embed_kernel(const int* __restrict__ idx,
                             const float* __restrict__ We,
                             const float* __restrict__ Wp,
                             unsigned short* __restrict__ xb,
                             int T, int C)
{
  const int token = blockIdx.x;              // b*T + t
  const int t  = token & (T - 1);
  const int id = idx[token];
  for (int c = threadIdx.x; c < C; c += blockDim.x)
    xb[(long)token * C + c] = f2bf(We[(long)id * C + c] + Wp[(long)t * C + c]);
}

// ---------------------------------------------------------------------------
// Wt[n][k] = bf16(W[k][n])   (weights pre-transposed for the BNT=1 GEMM path)
// ---------------------------------------------------------------------------
__global__ void cvt_transpose(const float* __restrict__ W,
                              unsigned short* __restrict__ Wt,
                              int K, int N)
{
  const long total = (long)K * N;
  for (long i = (long)blockIdx.x * blockDim.x + threadIdx.x; i < total;
       i += (long)gridDim.x * blockDim.x) {
    const long n = i / K, k = i % K;
    Wt[i] = f2bf(W[k * (long)N + n]);
  }
}

// ---------------------------------------------------------------------------
// Causal softmax over fp32 scores row; write bf16 probs (zeros above diagonal)
// ---------------------------------------------------------------------------
__global__ void softmax_causal(const float* __restrict__ wei,
                               unsigned short* __restrict__ pb, int T)
{
  __shared__ float red[256];
  const int t = blockIdx.x, b = blockIdx.y, tid = threadIdx.x;
  const float* row = wei + ((long)b * T + t) * T;
  unsigned short* out = pb + ((long)b * T + t) * T;
  const int L = t + 1;

  float m = -__builtin_inff();
  for (int s = tid; s < L; s += 256) m = fmaxf(m, row[s]);
  red[tid] = m; __syncthreads();
  for (int off = 128; off > 0; off >>= 1) {
    if (tid < off) red[tid] = fmaxf(red[tid], red[tid + off]);
    __syncthreads();
  }
  m = red[0]; __syncthreads();

  float sum = 0.f;
  for (int s = tid; s < L; s += 256) sum += __expf(row[s] - m);
  red[tid] = sum; __syncthreads();
  for (int off = 128; off > 0; off >>= 1) {
    if (tid < off) red[tid] += red[tid + off];
    __syncthreads();
  }
  const float inv = 1.f / red[0];

  for (int s = tid; s < T; s += 256)
    out[s] = (s < L) ? f2bf(__expf(row[s] - m) * inv) : (unsigned short)0;
}

// ---------------------------------------------------------------------------
extern "C" void kernel_launch(void* const* d_in, const int* in_sizes, int n_in,
                              void* d_out, int out_size, void* d_ws, size_t ws_size,
                              hipStream_t stream)
{
  const int V = 50257, C = 768, Bn = 4, T = 1024;
  const int BT = Bn * T;                     // 4096

  const int*   idx     = (const int*)  d_in[0];
  const float* W_embed = (const float*)d_in[1];
  const float* W_pos   = (const float*)d_in[2];
  const float* Wq      = (const float*)d_in[3];
  const float* Wk      = (const float*)d_in[4];
  const float* Wv      = (const float*)d_in[5];
  const float* W_lm    = (const float*)d_in[6];
  const float* b_lm    = (const float*)d_in[7];
  float* logits = (float*)d_out;

  // ---- carve workspace (256B aligned regions) ----
  char* p = (char*)d_ws;
  auto take = [&](size_t bytes) -> void* {
    void* r = (void*)p;
    p += (bytes + 255) & ~(size_t)255;
    return r;
  };
  unsigned short* xb   = (unsigned short*)take((size_t)BT * C * 2);
  unsigned short* wqt  = (unsigned short*)take((size_t)C * C * 2);
  unsigned short* wkt  = (unsigned short*)take((size_t)C * C * 2);
  unsigned short* wvt  = (unsigned short*)take((size_t)C * C * 2);
  unsigned short* wlmt = (unsigned short*)take((size_t)V * C * 2);
  unsigned short* qb   = (unsigned short*)take((size_t)BT * C * 2);
  unsigned short* kb   = (unsigned short*)take((size_t)BT * C * 2);
  unsigned short* vb   = (unsigned short*)take((size_t)BT * C * 2);
  float*          wei  = (float*)         take((size_t)Bn * T * T * 4);
  unsigned short* pb   = (unsigned short*)take((size_t)Bn * T * T * 2);
  unsigned short* ob   = (unsigned short*)take((size_t)BT * C * 2);

  const float scale = 1.0f / sqrtf((float)C);

  // 1) embedding -> bf16 x
  embed_kernel<<<BT, 256, 0, stream>>>(idx, W_embed, W_pos, xb, T, C);

  // 2) weight convert + transpose to bf16 [N][K]
  {
    const long nw = (long)C * C;
    const int g = (int)((nw + 255) / 256);
    cvt_transpose<<<g, 256, 0, stream>>>(Wq, wqt, C, C);
    cvt_transpose<<<g, 256, 0, stream>>>(Wk, wkt, C, C);
    cvt_transpose<<<g, 256, 0, stream>>>(Wv, wvt, C, C);
    const long nl = (long)C * V;
    cvt_transpose<<<(int)((nl + 255) / 256), 256, 0, stream>>>(W_lm, wlmt, C, V);
  }

  // 3) Q/K/V projections: [4096,768] = x @ W^T-form   (BNT=1, bf16 out)
  {
    dim3 grid(C / TN, BT / TM, 1);
    gemm_wmma<1, 1><<<grid, 256, 0, stream>>>(xb, wqt, qb, nullptr,
                                              BT, C, C, 0, 0, 0, 0.f);
    gemm_wmma<1, 1><<<grid, 256, 0, stream>>>(xb, wkt, kb, nullptr,
                                              BT, C, C, 0, 0, 0, 0.f);
    gemm_wmma<1, 1><<<grid, 256, 0, stream>>>(xb, wvt, vb, nullptr,
                                              BT, C, C, 0, 0, 0, 0.f);
  }

  // 4) attention scores: wei[b] = mask(q[b] @ k[b]^T) * scale  (batched, masked)
  {
    dim3 grid(T / TN, T / TM, Bn);
    gemm_wmma<1, 2><<<grid, 256, 0, stream>>>(qb, kb, wei, nullptr,
                                              T, T, C,
                                              (long)T * C, (long)T * C, (long)T * T,
                                              scale);
  }

  // 5) causal softmax -> bf16 probabilities
  {
    dim3 grid(T, Bn);
    softmax_causal<<<grid, 256, 0, stream>>>(wei, pb, T);
  }

  // 6) out[b] = P[b] @ V[b]   (BNT=0: V is [K=s][N=h] row-major; bf16 out)
  {
    dim3 grid(C / TN, T / TM, Bn);
    gemm_wmma<0, 1><<<grid, 256, 0, stream>>>(pb, vb, ob, nullptr,
                                              T, C, T,
                                              (long)T * T, (long)T * C, (long)T * C,
                                              0.f);
  }

  // 7) logits = out @ W_lm + b_lm   (fp32 out + bias, ragged N)
  {
    dim3 grid((V + TN - 1) / TN, BT / TM, 1);
    gemm_wmma<1, 0><<<grid, 256, 0, stream>>>(ob, wlmt, logits, b_lm,
                                              BT, V, C, 0, 0, 0, 0.f);
  }
}
